// NeuronDecoder_46420006535169
// MI455X (gfx1250) — matrix operand order repaired and checked by reading, not compiled, not run on hardware
//
#include <hip/hip_runtime.h>
#include <hip/hip_bf16.h>
#include <stdint.h>

// Problem constants (match reference)
#define B_   4
#define S_   512
#define N_   2048
#define E_   1024
#define QL_  32
#define H_   8
#define D_   2
#define VL_  32
#define SD_  (S_*D_)        // 1024
#define F_   (QL_*H_*D_)    // 512  (q feature dim)
#define G_   (VL_*H_*D_)    // 512  (vo feature dim / LN dim)
#define EPS_ 1e-5f
#define SCALE_ 0.17677669529663687f   // 1/sqrt(QL)

#define CHW_   512                    // attention column chunk width
#define CHSTR_ (32*CHW_)              // floats per LDS chunk block (32 rows)

typedef __attribute__((ext_vector_type(16))) __bf16 vbf16;
typedef __attribute__((ext_vector_type(8)))  float  v8f;
typedef __attribute__((ext_vector_type(4)))  unsigned int v4u;
typedef __attribute__((ext_vector_type(8)))  int v8i;
typedef __attribute__((ext_vector_type(4)))  int v4i;

union FragU { unsigned int u[8]; vbf16 v; };

__device__ __forceinline__ unsigned short f2bf(float f){
  unsigned int u = __float_as_uint(f);
  u += 0x7FFFu + ((u >> 16) & 1u);       // round-to-nearest-even
  return (unsigned short)(u >> 16);
}

// A-matrix (16x32 bf16, ISA 7.12.2): lane = M row; VGPR v holds K pair:
//   K = (v<4?0:16) + half*8 + (v%4)*2
__device__ __forceinline__ int kofsA(int v, int half){
  return ((v & 4) << 2) + half * 8 + ((v & 3) << 1);
}
// B-matrix (32x16 bf16): lane = N col; VGPR v holds K pair: same K as A
__device__ __forceinline__ int kofsB(int v, int half){
  return ((v & 4) << 2) + half * 8 + ((v & 3) << 1);
}

__device__ __forceinline__ vbf16 loadA_g(const unsigned short* row, int half){
  FragU f;
#pragma unroll
  for (int v = 0; v < 8; ++v)
    f.u[v] = *(const unsigned int*)(row + kofsA(v, half));
  return f.v;
}
__device__ __forceinline__ vbf16 loadB_g(const unsigned short* row, int half){
  FragU f;
#pragma unroll
  for (int v = 0; v < 8; ++v)
    f.u[v] = *(const unsigned int*)(row + kofsB(v, half));
  return f.v;
}
// Probs live in LDS as f32 -> convert to bf16 while building A fragment.
__device__ __forceinline__ vbf16 loadA_lds_f32(const float* row, int half){
  FragU f;
#pragma unroll
  for (int v = 0; v < 8; ++v){
    int k = kofsA(v, half);
    f.u[v] = (unsigned int)f2bf(row[k]) | ((unsigned int)f2bf(row[k+1]) << 16);
  }
  return f.v;
}

// ---------------------------------------------------------- TDM 2D tile load
// Builds a Tensor DMA Descriptor (ISA cdna5 ch.8) for a dense 2D f32 tile:
//   tile_d0 x tile_d1 elements, row stride `row_stride` (data_size units),
//   packed into LDS at byte offset lds_off. Groups 2/3 zero (dims unused).
// This toolchain (clang-23 / therock-10.0) exposes the 6-arg builtin:
//   (uint32x4 g0, int32x8 g1, int32x4, int32x4, int32x8, i32 cpol)
__device__ __forceinline__ void tdm_load_2d_f32(unsigned lds_off, const void* gptr,
                                                unsigned tensor_d0, unsigned tensor_d1,
                                                unsigned tile_d0, unsigned tile_d1,
                                                unsigned long long row_stride){
  unsigned long long ga = (unsigned long long)(uintptr_t)gptr;
  v4u g0;
  g0[0] = 1u;                                              // count=1, user desc
  g0[1] = lds_off;                                         // lds_addr (bytes)
  g0[2] = (unsigned)(ga & 0xFFFFFFFFull);                  // global_addr[31:0]
  g0[3] = (unsigned)((ga >> 32) & 0x01FFFFFFull)           // global_addr[56:32]
        | (2u << 30);                                      // type = 2 ("image")
  v8i g1;
  g1[0] = (int)(2u << 16);                                 // data_size=2 -> 4B
  g1[1] = (int)((tensor_d0 & 0xFFFFu) << 16);              // tensor_dim0[15:0]
  g1[2] = (int)((tensor_d0 >> 16) | ((tensor_d1 & 0xFFFFu) << 16));
  g1[3] = (int)((tensor_d1 >> 16) | (tile_d0 << 16));      // tile_dim0
  g1[4] = (int)(tile_d1 & 0xFFFFu);                        // tile_dim1 (dim2=0)
  g1[5] = (int)(unsigned)(row_stride & 0xFFFFFFFFull);     // dim0_stride[31:0]
  g1[6] = (int)(unsigned)((row_stride >> 32) & 0xFFFFull); // dim0_stride[47:32]
  g1[7] = 0;                                               // dim1_stride unused
  v4i z4 = {0, 0, 0, 0};
  v8i z8 = {0, 0, 0, 0, 0, 0, 0, 0};
  __builtin_amdgcn_tensor_load_to_lds(g0, g1, z4, z4, z8, 0);
}

// ---------------------------------------------------------------- pack kernels
__global__ void cvt_bf16_kernel(const float* __restrict__ src,
                                unsigned short* __restrict__ dst, int n){
  int i = blockIdx.x * blockDim.x + threadIdx.x;
  if (i < n) dst[i] = f2bf(src[i]);
}

// k_bf[b,h,n,ql] = keys[b,n,ql*H+h]
__global__ void pack_k_kernel(const float* __restrict__ keys,
                              unsigned short* __restrict__ k_bf){
  int i = blockIdx.x * blockDim.x + threadIdx.x;   // B*H*N*QL = 2^21
  int ql = i & 31;
  int n  = (i >> 5) & (N_ - 1);
  int h  = (i >> 16) & 7;
  int b  = i >> 19;
  k_bf[i] = f2bf(keys[((size_t)(b * N_ + n)) * (QL_ * H_) + ql * H_ + h]);
}

// v_bf[b,h,vl,n] = values[b,n,vl*H+h]
__global__ void pack_v_kernel(const float* __restrict__ values,
                              unsigned short* __restrict__ v_bf){
  int i = blockIdx.x * blockDim.x + threadIdx.x;   // B*H*VL*N = 2^21
  int n  = i & (N_ - 1);
  int vl = (i >> 11) & 31;
  int h  = (i >> 16) & 7;
  int b  = i >> 19;
  v_bf[i] = f2bf(values[((size_t)(b * N_ + n)) * (VL_ * H_) + vl * H_ + h]);
}

// --------------------------------------------------- Q projection (WMMA GEMM)
__global__ void q_gemm_kernel(const unsigned short* __restrict__ A,   // [2048][1024]
                              const unsigned short* __restrict__ Bm,  // [512][1024]
                              const float* __restrict__ bq,
                              unsigned short* __restrict__ q_bf){
  int tile = blockIdx.x * 4 + (threadIdx.x >> 5);   // 128 x 32 tiles
  int lane = threadIdx.x & 31;
  int half = lane >> 4, ln = lane & 15;
  int tm = tile >> 5;
  int tn = tile & 31;
  const unsigned short* aRow = A  + (size_t)(tm * 16 + ln) * E_;
  const unsigned short* bRow = Bm + (size_t)(tn * 16 + ln) * E_;
  v8f acc = {};
#pragma unroll 4
  for (int kc = 0; kc < E_ / 32; ++kc){
    vbf16 a = loadA_g(aRow + kc * 32, half);
    vbf16 b = loadB_g(bRow + kc * 32, half);
    acc = __builtin_amdgcn_wmma_f32_16x16x32_bf16(false, a, false, b,
                                                  (short)0, acc, false, false);
  }
  int f = tn * 16 + ln;
  float bias = bq[f];
  int d  = f >> 8;
  int ql = (f >> 3) & 31;
  int h  = f & 7;
#pragma unroll
  for (int r = 0; r < 8; ++r){
    int m  = tm * 16 + r + half * 8;
    int b_ = m >> 9, s = m & (S_ - 1);
    size_t dst = (((size_t)(b_ * H_ + h) * SD_) + (s * 2 + d)) * QL_ + ql;
    q_bf[dst] = f2bf(acc[r] + bias);
  }
}

// ------------------------------------------------------- attention (per 32 rows)
// grid.x = B*H * (SD/32) = 1024, block = 128 (4 waves), dyn LDS = 256.5 KB.
// LDS score buffer is chunk-major: 4 blocks of [32 rows][512 cols] f32 so each
// block is a dense TDM tile. The bias tensor (the 64 MB roofline term) is
// streamed by the Tensor Data Mover, double-buffered against the score WMMAs.
__global__ void attn_kernel(const unsigned short* __restrict__ q_bf, // [BH][SD][QL]
                            const unsigned short* __restrict__ k_bf, // [BH][N][QL]
                            const unsigned short* __restrict__ v_bf, // [BH][VL][N]
                            const float* __restrict__ ocs,           // [H][SD][N]
                            float* __restrict__ vo){                 // [B][S][G]
  extern __shared__ float smem[];
  float* sc   = smem;                 // 4 x [32][512]: bias -> scores -> exp
  float* red  = smem + 32 * N_;       // [32][4] partial reductions
  float* rinv = red + 32 * 4;         // [32] rowmax, then 1/rowsum

  int bh  = blockIdx.x >> 5;          // b*H+h
  int rb  = blockIdx.x & 31;
  int sd0 = rb * 32;
  int h   = bh & 7;
  int b   = bh >> 3;
  int w    = threadIdx.x >> 5;
  int lane = threadIdx.x & 31;
  int half = lane >> 4, ln = lane & 15;

  // Generic LDS pointers carry the LDS byte offset in their low 32 bits
  // (ISA 10.2 aperture mapping), which is what D#.lds_addr wants.
  unsigned lds_base = (unsigned)(size_t)(void*)smem;
  const float* ocs_base = ocs + ((size_t)h * SD_ + sd0) * N_;

  // Kick off chunk 0 bias DMA (one wave issues; TDM ignores EXEC).
  if (threadIdx.x == 0)
    tdm_load_2d_f32(lds_base, ocs_base, CHW_, 32, CHW_, 32, N_);

  // ---- Phase 1: scores = Q*K^T * scale - bias, chunk-pipelined vs TDM ----
  int rt = w >> 1;                    // wave row-tile (16 rows)
  int chh = w & 1;                    // wave half of the 512-col chunk
  const unsigned short* aRowQ =
      q_bf + ((size_t)bh * SD_ + sd0 + rt * 16 + ln) * QL_;
  vbf16 aq = loadA_g(aRowQ, half);    // K = QL = 32: one WMMA per 16x16 tile

  for (int c = 0; c < 4; ++c){
    if (threadIdx.x == 0){
      if (c + 1 < 4){                 // prefetch next chunk, keep depth 2
        tdm_load_2d_f32(lds_base + (unsigned)(c + 1) * (CHSTR_ * 4),
                        ocs_base + (size_t)(c + 1) * CHW_,
                        CHW_, 32, CHW_, 32, N_);
        __builtin_amdgcn_s_wait_tensorcnt((short)1);   // chunk c complete
      } else {
        __builtin_amdgcn_s_wait_tensorcnt((short)0);
      }
    }
    __syncthreads();                  // hand chunk c's bias to all waves

    float* scChunk = sc + c * CHSTR_;
#pragma unroll 2
    for (int ct = 0; ct < 16; ++ct){
      int ncol = chh * 256 + ct * 16 + ln;     // col within chunk
      int n = c * CHW_ + ncol;                 // global col
      const unsigned short* bRow = k_bf + ((size_t)bh * N_ + n) * QL_;
      vbf16 bf = loadB_g(bRow, half);
      v8f acc = {};
      acc = __builtin_amdgcn_wmma_f32_16x16x32_bf16(false, aq, false, bf,
                                                    (short)0, acc, false, false);
#pragma unroll
      for (int r = 0; r < 8; ++r){
        int lr = rt * 16 + r + half * 8;       // local row 0..31
        float* p = scChunk + lr * CHW_ + ncol;
        *p = acc[r] * SCALE_ - *p;             // bias already in LDS via TDM
      }
    }
  }
  __syncthreads();

  // ---- Phase 2: softmax stats (keep unnormalized exp; fold 1/sum later) ----
  {
    int row  = threadIdx.x >> 2;
    int part = threadIdx.x & 3;               // part == chunk id
    float* rp = sc + part * CHSTR_ + row * CHW_;
    float m = -3.402823466e38f;
    for (int i = 0; i < CHW_; ++i) m = fmaxf(m, rp[i]);
    red[row * 4 + part] = m;
    __syncthreads();
    if (threadIdx.x < 32){
      int t = threadIdx.x;
      rinv[t] = fmaxf(fmaxf(red[t*4], red[t*4+1]), fmaxf(red[t*4+2], red[t*4+3]));
    }
    __syncthreads();
    float rm = rinv[row];
    float ssum = 0.f;
    for (int i = 0; i < CHW_; ++i){
      float e = __expf(rp[i] - rm);
      rp[i] = e;
      ssum += e;
    }
    red[row * 4 + part] = ssum;
    __syncthreads();
    if (threadIdx.x < 32){
      int t = threadIdx.x;
      rinv[t] = 1.0f / (red[t*4] + red[t*4+1] + red[t*4+2] + red[t*4+3]);
    }
    __syncthreads();
  }

  // ---- Phase 3: PV = probs @ V, scatter into vo[b,s,h*64+d*32+vl] ----
  {
    int vt = w & 1;                           // wave vl-tile (rt reused)
    v8f acc = {};
    const unsigned short* vRow =
        v_bf + ((size_t)bh * VL_ + vt * 16 + ln) * N_;
    for (int cch = 0; cch < 4; ++cch){
      const float* aBase = sc + cch * CHSTR_ + (size_t)(rt * 16 + ln) * CHW_;
#pragma unroll 2
      for (int kk = 0; kk < CHW_ / 32; ++kk){
        vbf16 a  = loadA_lds_f32(aBase + kk * 32, half);
        vbf16 bf = loadB_g(vRow + cch * CHW_ + kk * 32, half);
        acc = __builtin_amdgcn_wmma_f32_16x16x32_bf16(false, a, false, bf,
                                                      (short)0, acc, false, false);
      }
    }
    int vl = vt * 16 + ln;
#pragma unroll
    for (int r = 0; r < 8; ++r){
      int lr = rt * 16 + r + half * 8;
      int sd = sd0 + lr;
      int s = sd >> 1, d = sd & 1;
      vo[((size_t)(b * S_ + s)) * G_ + h * (VL_ * D_) + d * VL_ + vl] =
          acc[r] * rinv[lr];
    }
  }
}

// ---------------------------------------------------------------- LayerNorm
__global__ void ln_kernel(const float* __restrict__ vo,
                          const float* __restrict__ gamma,
                          const float* __restrict__ beta,
                          unsigned short* __restrict__ xn_bf){
  int row  = blockIdx.x * 4 + (threadIdx.x >> 5);
  int lane = threadIdx.x & 31;
  const float* rp = vo + (size_t)row * G_;
  float x[16];
  float s = 0.f, sq = 0.f;
#pragma unroll
  for (int i = 0; i < 16; ++i){
    x[i] = rp[lane + 32 * i];
    s += x[i]; sq += x[i] * x[i];
  }
#pragma unroll
  for (int o = 16; o >= 1; o >>= 1){
    s  += __shfl_xor(s,  o, 32);
    sq += __shfl_xor(sq, o, 32);
  }
  float mean = s * (1.0f / G_);
  float var  = sq * (1.0f / G_) - mean * mean;
  float inv  = rsqrtf(var + EPS_);
#pragma unroll
  for (int i = 0; i < 16; ++i){
    int idx = lane + 32 * i;
    float xn = (x[i] - mean) * inv * gamma[idx] + beta[idx];
    xn_bf[(size_t)row * G_ + idx] = f2bf(xn);
  }
}

// ----------------------------------------------------- output GEMM (WMMA)
__global__ void out_gemm_kernel(const unsigned short* __restrict__ A,   // [2048][512]
                                const unsigned short* __restrict__ Bm,  // [1024][512]
                                const float* __restrict__ bo,
                                float* __restrict__ out){
  int tile = blockIdx.x * 4 + (threadIdx.x >> 5);   // 128 x 64 tiles
  int lane = threadIdx.x & 31;
  int half = lane >> 4, ln = lane & 15;
  int tm = tile >> 6;
  int tn = tile & 63;
  const unsigned short* aRow = A  + (size_t)(tm * 16 + ln) * G_;
  const unsigned short* bRow = Bm + (size_t)(tn * 16 + ln) * G_;
  v8f acc = {};
#pragma unroll 4
  for (int kc = 0; kc < G_ / 32; ++kc){
    vbf16 a = loadA_g(aRow + kc * 32, half);
    vbf16 b = loadB_g(bRow + kc * 32, half);
    acc = __builtin_amdgcn_wmma_f32_16x16x32_bf16(false, a, false, b,
                                                  (short)0, acc, false, false);
  }
  int o = tn * 16 + ln;
  float bias = bo[o];
#pragma unroll
  for (int r = 0; r < 8; ++r){
    int m = tm * 16 + r + half * 8;
    out[(size_t)m * E_ + o] = acc[r] + bias;
  }
}

// ------------------------------------------------------------------ launcher
extern "C" void kernel_launch(void* const* d_in, const int* in_sizes, int n_in,
                              void* d_out, int out_size, void* d_ws, size_t ws_size,
                              hipStream_t stream) {
  (void)in_sizes; (void)n_in; (void)out_size; (void)ws_size;
  const float* emb    = (const float*)d_in[0];
  const float* keys   = (const float*)d_in[1];
  const float* values = (const float*)d_in[2];
  const float* ocs    = (const float*)d_in[3];
  const float* Wq     = (const float*)d_in[4];
  const float* bq     = (const float*)d_in[5];
  const float* Wo     = (const float*)d_in[6];
  const float* bo     = (const float*)d_in[7];
  const float* gamma  = (const float*)d_in[8];
  const float* beta   = (const float*)d_in[9];
  float* out = (float*)d_out;

  // Workspace carve-up (bytes). Total ~22 MB.
  char* ws = (char*)d_ws;
  unsigned short* emb_bf = (unsigned short*)(ws);                       // 4 MB
  unsigned short* wq_bf  = (unsigned short*)(ws + (4u  << 20));         // 1 MB
  unsigned short* wo_bf  = (unsigned short*)(ws + (5u  << 20));         // 1 MB
  unsigned short* k_bf   = (unsigned short*)(ws + (6u  << 20));         // 4 MB
  unsigned short* v_bf   = (unsigned short*)(ws + (10u << 20));         // 4 MB
  unsigned short* q_bf   = (unsigned short*)(ws + (14u << 20));         // 2 MB
  float*          vo     = (float*)         (ws + (16u << 20));         // 4 MB
  unsigned short* xn_bf  = (unsigned short*)(ws + (20u << 20));         // 2 MB

  // 1) bf16 conversions of contiguous operands
  cvt_bf16_kernel<<<(B_*S_*E_ + 255)/256, 256, 0, stream>>>(emb, emb_bf, B_*S_*E_);
  cvt_bf16_kernel<<<(F_*E_   + 255)/256, 256, 0, stream>>>(Wq,  wq_bf,  F_*E_);
  cvt_bf16_kernel<<<(E_*G_   + 255)/256, 256, 0, stream>>>(Wo,  wo_bf,  E_*G_);

  // 2) repack K/V into WMMA-friendly layouts
  pack_k_kernel<<<(B_*H_*N_*QL_)/256, 256, 0, stream>>>(keys,   k_bf);
  pack_v_kernel<<<(B_*H_*VL_*N_)/256, 256, 0, stream>>>(values, v_bf);

  // 3) Q projection GEMM: 4096 16x16 tiles, 4 waves/block
  q_gemm_kernel<<<1024, 128, 0, stream>>>(emb_bf, wq_bf, bq, q_bf);

  // 4) attention: 1024 blocks * (32 rows x 2048 cols), 256.5 KB dynamic LDS,
  //    bias streamed via Tensor Data Mover
  size_t attn_smem = (size_t)(32 * N_ + 32 * 4 + 32) * sizeof(float);
  attn_kernel<<<1024, 128, attn_smem, stream>>>(q_bf, k_bf, v_bf, ocs, vo);

  // 5) LayerNorm -> bf16
  ln_kernel<<<512, 128, 0, stream>>>(vo, gamma, beta, xn_bf);

  // 6) output GEMM: 8192 16x16 tiles
  out_gemm_kernel<<<2048, 128, 0, stream>>>(xn_bf, wo_bf, bo, out);
}
